// Alignment_36386962932383
// MI455X (gfx1250) — compile-verified, matrix operand order checked by reading, not run on hardware
//
#include <hip/hip_runtime.h>

// ---------------------------------------------------------------------------
// Types for CDNA5 WMMA (gfx1250, wave32)
// ---------------------------------------------------------------------------
typedef _Float16 h16;
typedef __attribute__((ext_vector_type(16))) _Float16 v16h;
typedef __attribute__((ext_vector_type(8)))  _Float16 v8h;
typedef __attribute__((ext_vector_type(4)))  _Float16 v4h;
typedef __attribute__((ext_vector_type(8)))  float    v8f;

// ---------------------------------------------------------------------------
// CDNA5 async global->LDS path (ASYNCcnt-tracked). The builtin's params are
// int4* with address_space(1) (global src) / address_space(3) (LDS dst) --
// confirmed by the round-2 diagnostic ("int __vector(4) __device__ *").
// ---------------------------------------------------------------------------
#if __has_builtin(__builtin_amdgcn_global_load_async_to_lds_b128)
#define USE_ASYNC_LDS 1
typedef __attribute__((ext_vector_type(4))) int i4v;
typedef __attribute__((address_space(1))) i4v g_i4v;
typedef __attribute__((address_space(3))) i4v l_i4v;
#define ASYNC_B128(gp, lp) \
    __builtin_amdgcn_global_load_async_to_lds_b128((g_i4v*)(gp), (l_i4v*)(lp), 0, 0)
__device__ __forceinline__ void wait_async0() {
#if __has_builtin(__builtin_amdgcn_s_wait_asynccnt)
    __builtin_amdgcn_s_wait_asynccnt(0);
#else
    asm volatile("s_wait_asynccnt 0" ::: "memory");
#endif
}
#else
#define USE_ASYNC_LDS 0
#endif

#define KSTEP   64
#define ASTRIDE 72                 // 64 halfs + 8 pad: 144B rows, 16B aligned
#define ABUF    (32 * ASTRIDE)
#define BBUF    (64 * ASTRIDE)

__device__ __forceinline__ float fast_tanh(float x) {
    // tanh(x) = 1 - 2/(exp(2x)+1): v_exp_f32 + v_rcp_f32, saturates correctly
    const float e = __expf(2.0f * x);
    return 1.0f - 2.0f * __builtin_amdgcn_rcpf(e + 1.0f);
}

// 16-bit WMMA operand layout (ISA 7.12.2): lanes 0-15 hold K={0..7,16..23},
// lanes 16-31 hold K={8..15,24..31} -> two contiguous 16B LDS chunks per lane.
__device__ __forceinline__ v16h load_frag(const h16* __restrict__ rowp, int hi) {
    v8h lo = *(const v8h*)(rowp + hi * 8);
    v8h hh = *(const v8h*)(rowp + 16 + hi * 8);
    v16h r;
#pragma unroll
    for (int i = 0; i < 8; ++i) { r[i] = lo[i]; r[i + 8] = hh[i]; }
    return r;
}

// Two chained WMMAs consuming a K=64 slice.
__device__ __forceinline__ void wmma_k64(const h16* __restrict__ arow,
                                         const h16* __restrict__ brow,
                                         int hi, v8f& acc) {
    v16h a0 = load_frag(arow, hi);
    v16h b0 = load_frag(brow, hi);
    acc = __builtin_amdgcn_wmma_f32_16x16x32_f16(false, a0, false, b0, (short)0, acc, false, false);
    v16h a1 = load_frag(arow + 32, hi);
    v16h b1 = load_frag(brow + 32, hi);
    acc = __builtin_amdgcn_wmma_f32_16x16x32_f16(false, a1, false, b1, (short)0, acc, false, false);
}

// ---------------------------------------------------------------------------
// Generic f16 WMMA GEMM: C[M,N] = op(A[M,K] @ BT[N,K]^T + bias)
// 256 threads = 8 waves -> 32x64 macro tile; KSTEP=64 (2 WMMA per sync);
// double-buffered LDS with software pipelining (async-to-LDS when available).
// ---------------------------------------------------------------------------
template<bool TANH, bool OUT_HALF>
__global__ void __launch_bounds__(256)
gemm_wmma(const h16* __restrict__ A, const h16* __restrict__ BT,
          void* __restrict__ Cv, const float* __restrict__ bias,
          int K, int lda, int ldbt, int ldc,
          long sA, long sBT, long sC)
{
    __shared__ h16 ldsA[2 * ABUF];
    __shared__ h16 ldsB[2 * BBUF];

    const int z = blockIdx.z;
    const int tid = threadIdx.x, lane = tid & 31, wave = tid >> 5;
    const int waveM = wave >> 2, waveN = wave & 3;
    const int hi = lane >> 4, l15 = lane & 15;
    const int m0 = blockIdx.y * 32, n0 = blockIdx.x * 64;

    const int ar = tid >> 3, ac = (tid & 7) * 8;   // A: 1 x b128 per thread
    const int br = tid >> 2, bc = (tid & 3) * 16;  // B: 2 x b128 per thread

    const h16* __restrict__ Ag = A  + (long)z * sA  + (long)(m0 + ar) * lda  + ac;
    const h16* __restrict__ Bg = BT + (long)z * sBT + (long)(n0 + br) * ldbt + bc;

    h16* const la = &ldsA[ar * ASTRIDE + ac];
    h16* const lb = &ldsB[br * ASTRIDE + bc];
    const h16* const afrag = &ldsA[(waveM * 16 + l15) * ASTRIDE];
    const h16* const bfrag = &ldsB[(waveN * 16 + l15) * ASTRIDE];

    const int nk = K >> 6;
    v8f acc = {};

#if USE_ASYNC_LDS
    ASYNC_B128(Ag, la);
    ASYNC_B128(Bg, lb);
    ASYNC_B128(Bg + 8, lb + 8);
    wait_async0();
    __syncthreads();
    for (int kt = 0; kt < nk; ++kt) {
        const int cur = kt & 1, nxt = cur ^ 1;
        if (kt + 1 < nk) {           // DMA next tile while computing current
            const int k = (kt + 1) << 6;
            ASYNC_B128(Ag + k,     la + nxt * ABUF);
            ASYNC_B128(Bg + k,     lb + nxt * BBUF);
            ASYNC_B128(Bg + k + 8, lb + nxt * BBUF + 8);
        }
        wmma_k64(afrag + cur * ABUF, bfrag + cur * BBUF, hi, acc);
        wait_async0();
        __syncthreads();
    }
#else
    uint4 ra  = *(const uint4*)Ag;
    uint4 rb0 = *(const uint4*)Bg;
    uint4 rb1 = *(const uint4*)(Bg + 8);
    *(uint4*)la = ra; *(uint4*)lb = rb0; *(uint4*)(lb + 8) = rb1;
    __syncthreads();
    for (int kt = 0; kt < nk; ++kt) {
        const int cur = kt & 1, nxt = cur ^ 1;
        if (kt + 1 < nk) {           // issue next loads early; no wait yet
            const int k = (kt + 1) << 6;
            ra  = *(const uint4*)(Ag + k);
            rb0 = *(const uint4*)(Bg + k);
            rb1 = *(const uint4*)(Bg + k + 8);
        }
        wmma_k64(afrag + cur * ABUF, bfrag + cur * BBUF, hi, acc);
        __syncthreads();
        if (kt + 1 < nk) {
            *(uint4*)(la + nxt * ABUF) = ra;
            *(uint4*)(lb + nxt * BBUF) = rb0;
            *(uint4*)(lb + nxt * BBUF + 8) = rb1;
            __syncthreads();
        }
    }
#endif

    const int n = n0 + waveN * 16 + l15;
    const float bv = bias ? bias[n] : 0.0f;
#pragma unroll
    for (int r = 0; r < 8; ++r) {
        const int m = m0 + waveM * 16 + r + hi * 8;  // C/D layout: VGPR r -> M=r / r+8
        float v = acc[r] + bv;
        if (TANH) v = fast_tanh(v);
        if (OUT_HALF) ((h16*)Cv)[(long)z * sC + (long)m * ldc + n] = (h16)v;
        else          ((float*)Cv)[(long)z * sC + (long)m * ldc + n] = v;
    }
}

// ---------------------------------------------------------------------------
// Fused compare + aggregate (275-GFLOP hot spot):
//   v[b, side*FF + f] = sum_l tanh( [X | aligned][b,l,:] @ W_G[:,f] )
// The 64-col strip of W_G^T (64x1024 f16 = 128KB) stays RESIDENT in LDS for
// the whole block (320KB WGP budget), reused across 8 batches x 8 L-chunks;
// only the small A tiles stream through a double buffer.
// ---------------------------------------------------------------------------
#define NBATCH  8
#define WSTRIDE 1032               // 1024 + 8 pad halfs; 2064B rows (16B mult)

__global__ void __launch_bounds__(256)
compare_aggregate(const h16* __restrict__ Pbf, const h16* __restrict__ Hbf,
                  const h16* __restrict__ betas, const h16* __restrict__ alphas,
                  const h16* __restrict__ WGT,   // [2048][1024] = W_G^T
                  float* __restrict__ vout)      // [128][4096]
{
    extern __shared__ h16 smem[];
    h16*   WGs    = smem;                          // 64 * WSTRIDE halfs
    h16*   Abuf   = smem + 64 * WSTRIDE;           // 2 * ABUF halfs
    float* colsum = (float*)(Abuf + 2 * ABUF);     // 64 floats

    const int side = blockIdx.z;
    const int b0   = blockIdx.y * NBATCH;
    const int n0   = blockIdx.x * 64;

    const int tid = threadIdx.x, lane = tid & 31, wave = tid >> 5;
    const int waveM = wave >> 2, waveN = wave & 3;
    const int hi = lane >> 4, l15 = lane & 15;
    const int ar = tid >> 3, ac = (tid & 7) * 8;

    // --- load resident W_G^T strip: rows n0..n0+63, K = 0..1023
    {
        const int r = tid >> 2, cb = (tid & 3) * 16;
        const h16* __restrict__ wsrc = WGT + (long)(n0 + r) * 1024;
        h16* __restrict__ wdst = WGs + r * WSTRIDE;
#pragma unroll 4
        for (int ki = 0; ki < 16; ++ki) {
            const int c = ki * 64 + cb;
            *(uint4*)&wdst[c]     = *(const uint4*)&wsrc[c];
            *(uint4*)&wdst[c + 8] = *(const uint4*)&wsrc[c + 8];
        }
    }
    __syncthreads();

    const h16* __restrict__ Xbase  = side ? Hbf    : Pbf;
    const h16* __restrict__ Albase = side ? alphas : betas;
    h16* const la = &Abuf[ar * ASTRIDE + ac];
    const h16* const afrag = &Abuf[(waveM * 16 + l15) * ASTRIDE];
    const h16* const bfrag = &WGs[(waveN * 16 + l15) * WSTRIDE];

    for (int bi = 0; bi < NBATCH; ++bi) {
        const int b = b0 + bi;
        const h16* __restrict__ Xp = Xbase  + (long)b * 256 * 512;
        const h16* __restrict__ Ap = Albase + (long)b * 256 * 512;
        if (tid < 64) colsum[tid] = 0.0f;

        float psum = 0.0f;
        for (int lb = 0; lb < 256; lb += 32) {     // 32-row chunks of L
            v8f acc = {};
            auto asrc = [&](int kt) -> const h16* {  // concat([X, aligned]) K source
                const h16* base = (kt < 8) ? Xp : Ap;
                return base + (long)(lb + ar) * 512 + ((kt << 6) & 511) + ac;
            };
#if USE_ASYNC_LDS
            ASYNC_B128(asrc(0), la);
            wait_async0(); __syncthreads();
            for (int kt = 0; kt < 16; ++kt) {
                const int cur = kt & 1, nxt = cur ^ 1;
                if (kt + 1 < 16) ASYNC_B128(asrc(kt + 1), la + nxt * ABUF);
                wmma_k64(afrag + cur * ABUF, bfrag + kt * 64, hi, acc);
                wait_async0(); __syncthreads();
            }
#else
            uint4 ra = *(const uint4*)asrc(0);
            *(uint4*)la = ra;
            __syncthreads();
            for (int kt = 0; kt < 16; ++kt) {
                const int cur = kt & 1, nxt = cur ^ 1;
                if (kt + 1 < 16) ra = *(const uint4*)asrc(kt + 1);
                wmma_k64(afrag + cur * ABUF, bfrag + kt * 64, hi, acc);
                __syncthreads();
                if (kt + 1 < 16) { *(uint4*)(la + nxt * ABUF) = ra; __syncthreads(); }
            }
#endif
#pragma unroll
            for (int r = 0; r < 8; ++r) psum += fast_tanh(acc[r]);  // row-sum epilogue
        }

        psum += __shfl_down(psum, 16, 32);   // fold M-halves (lanes 16-31 -> 0-15)
        __syncthreads();
        if (lane < 16) atomicAdd(&colsum[waveN * 16 + lane], psum);  // ds_add_f32
        __syncthreads();
        if (tid < 64)
            vout[(long)b * 4096 + side * 2048 + n0 + tid] = colsum[tid];
    }
}

// ---------------------------------------------------------------------------
// Row softmax over eijs[b,p,:]; emits attn (f16) and attn^T (f16).
// ---------------------------------------------------------------------------
__global__ void __launch_bounds__(256)
softmax_kernel(const float* __restrict__ scores,
               h16* __restrict__ attn, h16* __restrict__ attnT)
{
    const int b = blockIdx.y, p = blockIdx.x, t = threadIdx.x;  // L == 256
    const long rowoff = ((long)b * 256 + p) * 256;
    __shared__ float red[256];

    float x = scores[rowoff + t];
    red[t] = x; __syncthreads();
    for (int s = 128; s > 0; s >>= 1) {
        if (t < s) red[t] = fmaxf(red[t], red[t + s]);
        __syncthreads();
    }
    const float mx = red[0];
    __syncthreads();
    const float e = __expf(x - mx);
    red[t] = e; __syncthreads();
    for (int s = 128; s > 0; s >>= 1) {
        if (t < s) red[t] += red[t + s];
        __syncthreads();
    }
    const h16 a = (h16)(e * __builtin_amdgcn_rcpf(red[0]));
    attn [rowoff + t] = a;
    attnT[((long)b * 256 + t) * 256 + p] = a;
}

// ---------------------------------------------------------------------------
// Conversions: straight f32->f16 (vectorized) and LDS-tiled transpose
// ---------------------------------------------------------------------------
__global__ void cvt_f16(const float* __restrict__ s, h16* __restrict__ d, long n4) {
    long i  = (long)blockIdx.x * blockDim.x + threadIdx.x;
    long st = (long)gridDim.x * blockDim.x;
    for (; i < n4; i += st) {
        const float4 f = ((const float4*)s)[i];
        v4h o; o[0] = (h16)f.x; o[1] = (h16)f.y; o[2] = (h16)f.z; o[3] = (h16)f.w;
        ((v4h*)d)[i] = o;
    }
}

// d[b][c][r] = (f16) s[b][r][c]; R,C multiples of 32; coalesced both sides.
__global__ void __launch_bounds__(256)
cvt_t_f16(const float* __restrict__ s, h16* __restrict__ d, int R, int C) {
    __shared__ float tile[32][33];
    const long base = (long)blockIdx.z * R * C;
    const int r0 = blockIdx.y * 32, c0 = blockIdx.x * 32;
    const int t = threadIdx.x;
    const int lr = t >> 3, lc4 = (t & 7) * 4;

    const float* __restrict__ sp = s + base + (long)(r0 + lr) * C + c0 + lc4;
#pragma unroll
    for (int i = 0; i < 4; ++i) tile[lr][lc4 + i] = sp[i];
    __syncthreads();

    v4h o;
#pragma unroll
    for (int i = 0; i < 4; ++i) o[i] = (h16)tile[lc4 + i][lr];
    *(v4h*)(d + base + (long)(c0 + lr) * R + r0 + lc4) = o;
}

// ---------------------------------------------------------------------------
// Final 3-class head: out[b,:] = a2[b,:] @ W3 + b3
// ---------------------------------------------------------------------------
__global__ void __launch_bounds__(256)
head_kernel(const float* __restrict__ a2, const float* __restrict__ W3,
            const float* __restrict__ b3, float* __restrict__ out)
{
    const int b = blockIdx.x, t = threadIdx.x;
    float s0 = 0.f, s1 = 0.f, s2 = 0.f;
    for (int k = t; k < 2048; k += 256) {
        const float av = a2[(long)b * 2048 + k];
        s0 += av * W3[k * 3 + 0];
        s1 += av * W3[k * 3 + 1];
        s2 += av * W3[k * 3 + 2];
    }
    __shared__ float r[3][256];
    r[0][t] = s0; r[1][t] = s1; r[2][t] = s2; __syncthreads();
    for (int s = 128; s > 0; s >>= 1) {
        if (t < s) { r[0][t] += r[0][t + s]; r[1][t] += r[1][t + s]; r[2][t] += r[2][t + s]; }
        __syncthreads();
    }
    if (t < 3) out[b * 3 + t] = r[t][0] + b3[t];
}

// ---------------------------------------------------------------------------
// Host-side orchestration
// ---------------------------------------------------------------------------
extern "C" void kernel_launch(void* const* d_in, const int* in_sizes, int n_in,
                              void* d_out, int out_size, void* d_ws, size_t ws_size,
                              hipStream_t stream) {
    (void)in_sizes; (void)n_in; (void)out_size; (void)ws_size;
    const int B = 128, L = 256, E = 512, A = 256, FF = 2048;

    const float* premises   = (const float*)d_in[0];
    const float* hypotheses = (const float*)d_in[1];
    const float* W_F = (const float*)d_in[2];
    const float* W_G = (const float*)d_in[3];
    const float* W1  = (const float*)d_in[4];
    const float* b1  = (const float*)d_in[5];
    const float* W2  = (const float*)d_in[6];
    const float* b2  = (const float*)d_in[7];
    const float* W3  = (const float*)d_in[8];
    const float* b3  = (const float*)d_in[9];

    char* ws = (char*)d_ws;
    size_t off = 0;
    auto alloc = [&](size_t bytes) -> void* {
        void* p = ws + off;
        off += (bytes + 255) & ~(size_t)255;
        return p;
    };

    const long nPE = (long)B * L * E;
    h16*  Pbf    = (h16*)alloc(nPE * 2);
    h16*  Hbf    = (h16*)alloc(nPE * 2);
    h16*  PT     = (h16*)alloc(nPE * 2);                  // [B][E][L]
    h16*  HT     = (h16*)alloc(nPE * 2);                  // [B][E][L]
    h16*  WFT    = (h16*)alloc((size_t)A * E * 2);        // [A][E]
    h16*  WGT    = (h16*)alloc((size_t)FF * 2 * E * 2);   // [FF][1024]
    h16*  W1T    = (h16*)alloc((size_t)FF * 2 * FF * 2);  // [FF][4096]
    h16*  W2T    = (h16*)alloc((size_t)FF * FF * 2);      // [FF][2048]
    h16*  Fp     = (h16*)alloc((size_t)B * L * A * 2);
    h16*  Fh     = (h16*)alloc((size_t)B * L * A * 2);
    float* scores = (float*)alloc((size_t)B * L * L * 4);
    h16*  attn   = (h16*)alloc((size_t)B * L * L * 2);
    h16*  attnT  = (h16*)alloc((size_t)B * L * L * 2);
    h16*  betas  = (h16*)alloc(nPE * 2);
    h16*  alphas = (h16*)alloc(nPE * 2);
    float* vf    = (float*)alloc((size_t)B * 2 * FF * 4);
    h16*  vbf    = (h16*)alloc((size_t)B * 2 * FF * 2);
    float* a1f   = (float*)alloc((size_t)B * FF * 4);
    h16*  a1h    = (h16*)alloc((size_t)B * FF * 2);
    float* a2f   = (float*)alloc((size_t)B * FF * 4);

    // --- 1. precision conversion + operand transposes
    cvt_f16<<<2048, 256, 0, stream>>>(premises,   Pbf, nPE / 4);
    cvt_f16<<<2048, 256, 0, stream>>>(hypotheses, Hbf, nPE / 4);
    cvt_t_f16<<<dim3(16, 8, B), 256, 0, stream>>>(premises,   PT, L, E);
    cvt_t_f16<<<dim3(16, 8, B), 256, 0, stream>>>(hypotheses, HT, L, E);
    cvt_t_f16<<<dim3(8, 16, 1), 256, 0, stream>>>(W_F, WFT, E, A);
    cvt_t_f16<<<dim3(64, 32, 1), 256, 0, stream>>>(W_G, WGT, 2 * E, FF);
    cvt_t_f16<<<dim3(64, 128, 1), 256, 0, stream>>>(W1, W1T, 2 * FF, FF);
    cvt_t_f16<<<dim3(64, 64, 1), 256, 0, stream>>>(W2, W2T, FF, FF);

    // --- 2. projections: F = tanh(X @ W_F), flat M = B*L
    gemm_wmma<true, true><<<dim3(A / 64, (B * L) / 32, 1), 256, 0, stream>>>(
        Pbf, WFT, Fp, nullptr, E, E, E, A, 0, 0, 0);
    gemm_wmma<true, true><<<dim3(A / 64, (B * L) / 32, 1), 256, 0, stream>>>(
        Hbf, WFT, Fh, nullptr, E, E, E, A, 0, 0, 0);

    // --- 3. alignment scores eijs = F_p @ F_h^T (batched)
    gemm_wmma<false, false><<<dim3(L / 64, L / 32, B), 256, 0, stream>>>(
        Fp, Fh, scores, nullptr, A, A, A, L,
        (long)L * A, (long)L * A, (long)L * L);

    // --- 4. softmax over h; write attn and attn^T
    softmax_kernel<<<dim3(L, B), 256, 0, stream>>>(scores, attn, attnT);

    // --- 5. soft alignments
    gemm_wmma<false, true><<<dim3(E / 64, L / 32, B), 256, 0, stream>>>(
        attn, HT, betas, nullptr, L, L, L, E,
        (long)L * L, (long)E * L, (long)L * E);
    gemm_wmma<false, true><<<dim3(E / 64, L / 32, B), 256, 0, stream>>>(
        attnT, PT, alphas, nullptr, L, L, L, E,
        (long)L * L, (long)E * L, (long)L * E);

    // --- 6. fused compare + length-sum aggregate (W_G strip resident in LDS)
    const size_t cmp_lds = (64 * WSTRIDE + 2 * ABUF) * sizeof(h16) + 64 * sizeof(float);
    compare_aggregate<<<dim3(FF / 64, B / NBATCH, 2), 256, cmp_lds, stream>>>(
        Pbf, Hbf, betas, alphas, WGT, vf);

    // --- 7. classifier MLP
    cvt_f16<<<512, 256, 0, stream>>>(vf, vbf, (long)B * 2 * FF / 4);
    gemm_wmma<true, false><<<dim3(FF / 64, B / 32, 1), 256, 0, stream>>>(
        vbf, W1T, a1f, b1, 2 * FF, 2 * FF, 2 * FF, FF, 0, 0, 0);
    cvt_f16<<<256, 256, 0, stream>>>(a1f, a1h, (long)B * FF / 4);
    gemm_wmma<true, false><<<dim3(FF / 64, B / 32, 1), 256, 0, stream>>>(
        a1h, W2T, a2f, b2, FF, FF, FF, FF, 0, 0, 0);

    // --- 8. 3-class head
    head_kernel<<<dim3(B), 256, 0, stream>>>(a2f, W3, b3, (float*)d_out);
}
